// MemoryCube_65755949301985
// MI455X (gfx1250) — matrix operand order repaired and checked by reading, not compiled, not run on hardware
//
#include <hip/hip_runtime.h>
#include <hip/hip_bf16.h>

typedef __attribute__((ext_vector_type(16))) _Float16 v16h;
typedef __attribute__((ext_vector_type(8)))  _Float16 v8h;
typedef __attribute__((ext_vector_type(4)))  _Float16 v4h;
typedef __attribute__((ext_vector_type(8)))  float    v8f;
typedef __attribute__((ext_vector_type(4)))  int      v4i;

#define B_ROWS    4096
#define DK        512
#define NKEYS     65536
#define NSEG      16
#define SEG_COLS  (NKEYS / NSEG)          // 4096
#define TILE_COLS 16
#define ITERS     (SEG_COLS / TILE_COLS)  // 256
#define RPB       128                     // rows per block (8 waves x 16)
#define WAVES     8
#define BLK       (WAVES * 32)            // 256 threads
#define TOPK      8
#define EPSF      1e-8f
#define BSTRIDE   520                     // skewed LDS row stride (halves)

#define OUT_OFF   0
#define CONF_OFF  ((size_t)B_ROWS * DK)   // 2097152
#define IDX_OFF   (CONF_OFF + 1)
#define ATTN_OFF  (IDX_OFF + (size_t)B_ROWS * TOPK)

// ---- CDNA5 async global->LDS staging (guarded; sync-copy fallback) --------
#if defined(__gfx1250__) && __has_builtin(__builtin_amdgcn_global_load_async_to_lds_b128)
#define USE_ASYNC_LDS 1
#else
#define USE_ASYNC_LDS 0
#endif

// scheduling-group barrier: pin DS-read / WMMA interleave (no-op if absent)
#if defined(__gfx1250__) && __has_builtin(__builtin_amdgcn_sched_group_barrier)
#define SGB(mask, size) __builtin_amdgcn_sched_group_barrier(mask, size, 0)
#else
#define SGB(mask, size)
#endif

__device__ __forceinline__ void async_copy16(const _Float16* gsrc, _Float16* ldst) {
#if USE_ASYNC_LDS
    // probe-verified signature: (int4 AS1*, int4 AS3*, imm offset, imm cpol)
    __builtin_amdgcn_global_load_async_to_lds_b128(
        (__attribute__((address_space(1))) v4i*)gsrc,
        (__attribute__((address_space(3))) v4i*)ldst, 0, 0);
#else
    *(v8h*)ldst = *(const v8h*)gsrc;
#endif
}

__device__ __forceinline__ void async_fence() {
#if USE_ASYNC_LDS
#if __has_builtin(__builtin_amdgcn_s_wait_asynccnt)
    __builtin_amdgcn_s_wait_asynccnt(0);
#else
    asm volatile("s_wait_asynccnt 0x0" ::: "memory");
#endif
#endif
}

__device__ __forceinline__ float wave_sum(float v) {
#pragma unroll
    for (int o = 16; o > 0; o >>= 1) v += __shfl_xor(v, o, 32);
    return v;
}

// sorted-descending top-8 insertion (arrays stay in registers under full unroll)
__device__ __forceinline__ void top8_insert(float val, int idx, float (&tv)[8], int (&ti)[8]) {
    if (val <= tv[7]) return;
    float cv = val; int ci = idx;
#pragma unroll
    for (int t = 0; t < 8; ++t) {
        if (cv > tv[t]) {
            float fv = tv[t]; int fi = ti[t];
            tv[t] = cv; ti[t] = ci;
            cv = fv; ci = fi;
        }
    }
}

// B fragment: lane holds K = ks*32 + hl*8 .. +7 and ks*32 + 16 + hl*8 .. +7 of column l15
__device__ __forceinline__ v16h load_bfrag(const _Float16* bbase, int ks) {
    const v8h b0 = *(const v8h*)(bbase + ks * 32);
    const v8h b1 = *(const v8h*)(bbase + ks * 32 + 16);
    return __builtin_shufflevector(b0, b1,
        0, 1, 2, 3, 4, 5, 6, 7, 8, 9, 10, 11, 12, 13, 14, 15);
}

// ---------------------------------------------------------------------------
// Kernel 1: keys_n = l2norm(mem_keys) -> f16, one wave per key row
// ---------------------------------------------------------------------------
__global__ void __launch_bounds__(256) k_norm_keys(const float* __restrict__ keys,
                                                   _Float16* __restrict__ keysH) {
    const int lane = threadIdx.x & 31;
    const int wave = threadIdx.x >> 5;
    const int row  = blockIdx.x * 8 + wave;
    const float* kr = keys + (size_t)row * DK;

    float4 v[4];
    float ss = 0.f;
#pragma unroll
    for (int t = 0; t < 4; ++t) {
        v[t] = *(const float4*)(kr + lane * 4 + 128 * t);
        ss += v[t].x * v[t].x + v[t].y * v[t].y + v[t].z * v[t].z + v[t].w * v[t].w;
    }
    ss = wave_sum(ss);
    const float inv = 1.f / (sqrtf(ss) + EPSF);

    _Float16* outr = keysH + (size_t)row * DK;
#pragma unroll
    for (int t = 0; t < 4; ++t) {
        v4h h;
        h[0] = (_Float16)(v[t].x * inv);
        h[1] = (_Float16)(v[t].y * inv);
        h[2] = (_Float16)(v[t].z * inv);
        h[3] = (_Float16)(v[t].w * inv);
        *(v4h*)(outr + lane * 4 + 128 * t) = h;
    }
}

// ---------------------------------------------------------------------------
// Kernel 2: qn = l2n( 0.7*l2n(q) + 0.3*(spike @ W^T + b) ) -> f16
// one wave per batch row; also zero-initializes the conf output slot
// ---------------------------------------------------------------------------
__global__ void __launch_bounds__(32) k_fuse_q(const float* __restrict__ q,
                                               const float* __restrict__ spike,
                                               const float* __restrict__ W,
                                               const float* __restrict__ bias,
                                               _Float16* __restrict__ qnH,
                                               float* __restrict__ out) {
    __shared__ float sp[DK];
    const int row = blockIdx.x, lane = threadIdx.x;
    const size_t rb = (size_t)row * DK;

    for (int d = lane; d < DK; d += 32) sp[d] = spike[rb + d];

    float qv[16];
    float ss = 0.f;
#pragma unroll
    for (int t = 0; t < 16; ++t) {
        const int d = lane + 32 * t;
        qv[t] = q[rb + d];
        ss += qv[t] * qv[t];
    }
    ss = wave_sum(ss);
    const float inv = 1.f / (sqrtf(ss) + EPSF);
    __syncthreads();

    float fused[16];
    float ss2 = 0.f;
#pragma unroll
    for (int t = 0; t < 16; ++t) {
        const int j = lane + 32 * t;
        const float* wr = W + (size_t)j * DK;
        float e = bias[j];
        for (int k = 0; k < DK; k += 4) {
            const float4 wv = *(const float4*)(wr + k);
            e += sp[k] * wv.x + sp[k + 1] * wv.y + sp[k + 2] * wv.z + sp[k + 3] * wv.w;
        }
        const float f = 0.7f * qv[t] * inv + 0.3f * e;
        fused[t] = f;
        ss2 += f * f;
    }
    ss2 = wave_sum(ss2);
    const float inv2 = 1.f / (sqrtf(ss2) + EPSF);
#pragma unroll
    for (int t = 0; t < 16; ++t)
        qnH[rb + lane + 32 * t] = (_Float16)(fused[t] * inv2);

    if (row == 0 && lane == 0) out[CONF_OFF] = 0.f;
}

// ---------------------------------------------------------------------------
// Kernel 3: sims = qn @ keys^T (f16 WMMA), fused per-row top-8.
// 128 rows/block (8 waves x 16 rows), 16-col key tile double-buffered in LDS
// via async global->LDS DMA, A fragments hoisted to registers, B fragments
// rotated at depth 3 with a sched_group_barrier-pinned DS-read/WMMA schedule.
// ---------------------------------------------------------------------------
__global__ void __launch_bounds__(BLK) k_sims_topk(const _Float16* __restrict__ keysH,
                                                   const _Float16* __restrict__ qnH,
                                                   float* __restrict__ pvals,
                                                   int* __restrict__ pidx) {
    __shared__ _Float16 Bs[2][TILE_COLS][BSTRIDE]; // 2 x 16 x 512 f16, skewed rows
    __shared__ float    Ss[RPB][20];               // sims tile, padded vs bank conflicts
    __shared__ float    Mv[RPB][16];               // merge candidates (vals)
    __shared__ int      Mi[RPB][16];               // merge candidates (idx)

    const int tid    = threadIdx.x;
    const int lane   = tid & 31;
    const int w      = tid >> 5;
    const int rowblk = blockIdx.x;
    const int seg    = blockIdx.y;
    const int hl     = lane >> 4;
    const int l15    = lane & 15;

    // ---- hoist this wave's A fragments (16 rows x 512 K) into registers ----
    // A lane layout (16-bit 16x32): elems 0-7 = K=k0+hl*8.., elems 8-15 = K=k0+16+hl*8..
    const _Float16* aRow = qnH + (size_t)(rowblk * RPB + w * 16 + l15) * DK;
    v16h areg[16];
#pragma unroll
    for (int ks = 0; ks < 16; ++ks) {
        const int k0 = ks * 32;
        const v8h a0 = *(const v8h*)(aRow + k0 + hl * 8);
        const v8h a1 = *(const v8h*)(aRow + k0 + 16 + hl * 8);
        areg[ks] = __builtin_shufflevector(a0, a1,
            0, 1, 2, 3, 4, 5, 6, 7, 8, 9, 10, 11, 12, 13, 14, 15);
    }

    float tv[8]; int ti[8];
#pragma unroll
    for (int j = 0; j < 8; ++j) { tv[j] = -3.4e38f; ti[j] = 0; }

    const int seg_base = seg * SEG_COLS;

    // ---- per-thread staging addresses (4 chunks, stride 4 key rows) --------
    const int r0 = tid >> 6;              // 0..3
    const int k8 = (tid & 63) * 8;
    _Float16* lds0 = &Bs[0][r0][k8];
    _Float16* lds1 = &Bs[1][r0][k8];
    // running global pointer for the tile currently being staged
    const _Float16* gcur = keysH + (size_t)(seg_base + r0) * DK + k8;

    // stage B tile for iteration 0 into buffer 0
    async_copy16(gcur,           lds0);
    async_copy16(gcur + 4 * DK,  lds0 + 4 * BSTRIDE);
    async_copy16(gcur + 8 * DK,  lds0 + 8 * BSTRIDE);
    async_copy16(gcur + 12 * DK, lds0 + 12 * BSTRIDE);
    gcur += (size_t)TILE_COLS * DK;
    async_fence();
    __syncthreads();

    for (int it = 0; it < ITERS; ++it) {
        const int buf = it & 1;
        const int colbase = seg_base + it * TILE_COLS;

        // issue async staging of next tile (overlaps the WMMAs below)
        if (it + 1 < ITERS) {
            _Float16* lb = (buf ^ 1) ? lds1 : lds0;
            async_copy16(gcur,           lb);
            async_copy16(gcur + 4 * DK,  lb + 4 * BSTRIDE);
            async_copy16(gcur + 8 * DK,  lb + 8 * BSTRIDE);
            async_copy16(gcur + 12 * DK, lb + 12 * BSTRIDE);
            gcur += (size_t)TILE_COLS * DK;
        }

        // 16x16 tile: 16 chained WMMAs, A from registers, B frags rotated
        // at depth 3 (fragment f loaded 3 WMMAs before its consumer)
        const _Float16* bbase = &Bs[buf][l15][hl * 8];
        v8f c = {};
        v16h bb[3];
        bb[0] = load_bfrag(bbase, 0);
        bb[1] = load_bfrag(bbase, 1);
        bb[2] = load_bfrag(bbase, 2);
#pragma unroll
        for (int ks = 0; ks < 16; ++ks) {
            c = __builtin_amdgcn_wmma_f32_16x16x32_f16(
                    false, areg[ks], false, bb[ks % 3], (short)0, c, false, false);
            if (ks + 3 < 16) bb[ks % 3] = load_bfrag(bbase, ks + 3);
        }

        // pin the schedule: 6 DS-reads | (1 WMMA, 2 DS-reads) x13 | 3 WMMAs
        SGB(0x100, 6);
#pragma unroll
        for (int g = 0; g < 13; ++g) { SGB(0x008, 1); SGB(0x100, 2); }
        SGB(0x008, 3);

        // C layout: lane l, vgpr v -> M = v + hl*8, N = l15 (within this wave's rows)
#pragma unroll
        for (int v = 0; v < 8; ++v)
            Ss[w * 16 + v + hl * 8][l15] = c[v];
        __syncthreads();

        // top-k update: 2 threads per row, 8 columns each
        {
            const int r = tid >> 1, s = tid & 1;
#pragma unroll
            for (int qq = 0; qq < 8; ++qq) {
                const int cc = s + 2 * qq;
                top8_insert(Ss[r][cc], colbase + cc, tv, ti);
            }
        }

        // next tile's async data must be in LDS, and everyone must be done with
        // this Ss read and with the buffer we overwrite next iteration
        async_fence();
        __syncthreads();
    }

    // ---- block-level merge: 2 partials per row -> top-8 per (row, seg) ----
    {
        const int r = tid >> 1, s = tid & 1;
#pragma unroll
        for (int j = 0; j < 8; ++j) { Mv[r][s * 8 + j] = tv[j]; Mi[r][s * 8 + j] = ti[j]; }
    }
    __syncthreads();

    if (tid < RPB) {
        float bv[8]; int bi[8];
#pragma unroll
        for (int j = 0; j < 8; ++j) { bv[j] = -3.4e38f; bi[j] = 0; }
        for (int cc = 0; cc < 16; ++cc)
            top8_insert(Mv[tid][cc], Mi[tid][cc], bv, bi);
        const size_t base = ((size_t)(rowblk * RPB + tid) * NSEG + seg) * 8;
#pragma unroll
        for (int j = 0; j < 8; ++j) { pvals[base + j] = bv[j]; pidx[base + j] = bi[j]; }
    }
}

// ---------------------------------------------------------------------------
// Kernel 4: merge segment partials, softmax, entropy->conf, gather+combine
// one wave per batch row
// ---------------------------------------------------------------------------
__global__ void __launch_bounds__(32) k_finalize(const float* __restrict__ pvals,
                                                 const int* __restrict__ pidx,
                                                 const float* __restrict__ memv,
                                                 float* __restrict__ out) {
    __shared__ float cv[NSEG * 8];
    __shared__ int   ci[NSEG * 8];
    __shared__ float tv8[8];
    __shared__ int   ti8[8];

    const int row = blockIdx.x, l = threadIdx.x;
    const size_t base = (size_t)row * (NSEG * 8);
#pragma unroll
    for (int t = 0; t < (NSEG * 8) / 32; ++t) {
        cv[l + 32 * t] = pvals[base + l + 32 * t];
        ci[l + 32 * t] = pidx[base + l + 32 * t];
    }
    __syncthreads();

    if (l == 0) {
        float bv[8]; int bi[8];
#pragma unroll
        for (int j = 0; j < 8; ++j) { bv[j] = -3.4e38f; bi[j] = 0; }
        for (int cc = 0; cc < NSEG * 8; ++cc) top8_insert(cv[cc], ci[cc], bv, bi);
#pragma unroll
        for (int j = 0; j < 8; ++j) { tv8[j] = bv[j]; ti8[j] = bi[j]; }
    }
    __syncthreads();

    // softmax(top_vals / 0.1), computed redundantly per lane
    float a[8];
    const float mx = tv8[0];  // sorted descending
    float s = 0.f;
#pragma unroll
    for (int k = 0; k < 8; ++k) { a[k] = __expf((tv8[k] - mx) * 10.f); s += a[k]; }
    const float isum = 1.f / s;
#pragma unroll
    for (int k = 0; k < 8; ++k) a[k] *= isum;

    if (l == 0) {
        float H = 0.f;
#pragma unroll
        for (int k = 0; k < 8; ++k) { const float p = fmaxf(a[k], EPSF); H -= p * __logf(p); }
        const float conf = (1.f - H / (logf(8.f) + EPSF)) * (1.f / (float)B_ROWS);
        atomicAdd(out + CONF_OFF, conf);
    }
    if (l < 8) {
        out[IDX_OFF + (size_t)row * 8 + l]  = (float)ti8[l];
        out[ATTN_OFF + (size_t)row * 8 + l] = a[l];
    }

    for (int d = l; d < DK; d += 32) {
        float acc = 0.f;
#pragma unroll
        for (int k = 0; k < 8; ++k)
            acc += a[k] * memv[(size_t)ti8[k] * DK + d];
        out[OUT_OFF + (size_t)row * DK + d] = acc;
    }
}

// ---------------------------------------------------------------------------
extern "C" void kernel_launch(void* const* d_in, const int* in_sizes, int n_in,
                              void* d_out, int out_size, void* d_ws, size_t ws_size,
                              hipStream_t stream) {
    (void)in_sizes; (void)n_in; (void)out_size; (void)ws_size;

    const float* q     = (const float*)d_in[0];
    const float* spike = (const float*)d_in[1];
    const float* keys  = (const float*)d_in[2];
    const float* memv  = (const float*)d_in[3];
    const float* W     = (const float*)d_in[4];
    const float* bias  = (const float*)d_in[5];
    float* out = (float*)d_out;

    char* ws = (char*)d_ws;
    _Float16* keysH = (_Float16*)ws;                                   // 64 MB (L2-resident)
    _Float16* qnH   = (_Float16*)(ws + (size_t)NKEYS * DK * 2);        // 4 MB
    float*    pvals = (float*)(ws + (size_t)NKEYS * DK * 2
                                  + (size_t)B_ROWS * DK * 2);          // 2 MB
    int*      pidx  = (int*)((char*)pvals + (size_t)B_ROWS * NSEG * 8 * 4); // 2 MB

    k_norm_keys<<<NKEYS / 8, 256, 0, stream>>>(keys, keysH);
    k_fuse_q<<<B_ROWS, 32, 0, stream>>>(q, spike, W, bias, qnH, out);
    k_sims_topk<<<dim3(B_ROWS / RPB, NSEG), BLK, 0, stream>>>(keysH, qnH, pvals, pidx);
    k_finalize<<<B_ROWS, 32, 0, stream>>>(pvals, pidx, memv, out);
}
